// TokenPruner_44272522887697
// MI455X (gfx1250) — matrix operand order repaired and checked by reading, not compiled, not run on hardware
//
#include <hip/hip_runtime.h>
#include <hip/hip_bf16.h>

typedef _Float16 v16h __attribute__((ext_vector_type(16)));
typedef _Float16 v8h  __attribute__((ext_vector_type(8)));
typedef float    v8f  __attribute__((ext_vector_type(8)));

constexpr int B_ = 8;
constexpr int S_ = 2561;
constexpr int D_ = 1024;
constexpr int P_ = 2048;   // NUM_PATCHES
constexpr int T_ = 512;    // task tokens = S - 1 - P

__device__ __forceinline__ v8f zero_v8f() {
  v8f z;
#pragma unroll
  for (int i = 0; i < 8; i++) z[i] = 0.0f;
  return z;
}

// ---- WMMA fragment loaders (ISA 7.12.2 layouts, NT GEMM: both operands row-major, K contiguous) ----
// A-matrix 16x32 f16: lane m (0..15): halves[0..7]=K0..7, [8..15]=K16..23 ; lane m+16: K8..15, K24..31
__device__ __forceinline__ v16h load_fragA(const _Float16* __restrict__ base, int ld,
                                           int row_base, int kk, int lane) {
  int r = row_base + (lane & 15);
  int h = lane >> 4;
  const _Float16* p = base + (size_t)r * ld + kk + 8 * h;
  v8h lo = *(const v8h*)p;
  v8h hi = *(const v8h*)(p + 16);
  return __builtin_shufflevector(lo, hi, 0,1,2,3,4,5,6,7,8,9,10,11,12,13,14,15);
}

// B-matrix 32x16 f16: lane n (0..15): halves[0..7]=K0..7, [8..15]=K16..23 of column n ; lane n+16: K8..15, K24..31
__device__ __forceinline__ v16h load_fragB(const _Float16* __restrict__ base, int ld,
                                           int col_base, int kk, int lane) {
  int c = col_base + (lane & 15);
  int h = lane >> 4;
  const _Float16* p = base + (size_t)c * ld + kk + 8 * h;
  v8h lo = *(const v8h*)p;
  v8h hi = *(const v8h*)(p + 16);
  return __builtin_shufflevector(lo, hi, 0,1,2,3,4,5,6,7,8,9,10,11,12,13,14,15);
}

// ---- RMSNorm: 256 threads per row of 1024, f32 or f16 input -> f16 output ----
template <typename TIn>
__global__ void rmsnorm_kernel(const TIn* __restrict__ src, _Float16* __restrict__ dst,
                               int src_batch_rows, int src_row_off, int dst_batch_rows) {
  int r = blockIdx.x, b = blockIdx.y, tid = threadIdx.x;
  const TIn* srow = src + ((size_t)b * src_batch_rows + src_row_off + r) * D_;
  _Float16* drow = dst + ((size_t)b * dst_batch_rows + r) * D_;
  float v[4];
#pragma unroll
  for (int i = 0; i < 4; i++) v[i] = (float)srow[tid * 4 + i];
  float ss = v[0]*v[0] + v[1]*v[1] + v[2]*v[2] + v[3]*v[3];
#pragma unroll
  for (int m = 1; m < 32; m <<= 1) ss += __shfl_xor(ss, m, 32);
  __shared__ float red[8];
  if ((tid & 31) == 0) red[tid >> 5] = ss;
  __syncthreads();
  float tot = 0.0f;
#pragma unroll
  for (int i = 0; i < 8; i++) tot += red[i];
  float rs = rsqrtf(tot * (1.0f / D_) + 1e-6f);
#pragma unroll
  for (int i = 0; i < 4; i++) drow[tid * 4 + i] = (_Float16)(v[i] * rs);
}

// ---- Transpose tn (B,T,D) -> tnT (B,D,T), coalesced writes ----
__global__ void transpose_kernel(const _Float16* __restrict__ src, _Float16* __restrict__ dst) {
  int lin = blockIdx.x * blockDim.x + threadIdx.x;  // over D*T per batch
  int b = blockIdx.y;
  int t = lin % T_;
  int d = lin / T_;
  dst[(size_t)b * D_ * T_ + (size_t)d * T_ + t] = src[(size_t)b * T_ * D_ + (size_t)t * D_ + d];
}

// ---- NT GEMM: C[M,N] = scale * A(MxK) * B(NxK)^T ; 8 waves, block tile 16x512, wave tile 16x64 ----
template <typename TOut>
__global__ void __launch_bounds__(256, 1)
gemm_nt_kernel(const _Float16* __restrict__ A, int lda,
               const _Float16* __restrict__ Bm, int ldb,
               TOut* __restrict__ C, int ldc,
               int K, float scale,
               size_t a_batch, size_t b_batch, size_t c_batch) {
  int lane = threadIdx.x & 31, w = threadIdx.x >> 5;
  int b = blockIdx.z;
  const _Float16* Ab = A + (size_t)b * a_batch;
  const _Float16* Bb = Bm + (size_t)b * b_batch;
  TOut* Cb = C + (size_t)b * c_batch;
  int row_base = blockIdx.x * 16;
  int col0 = blockIdx.y * 512 + w * 64;
  v8f acc[4];
#pragma unroll
  for (int j = 0; j < 4; j++) acc[j] = zero_v8f();
#pragma unroll 1
  for (int kk = 0; kk < K; kk += 32) {
    v16h a = load_fragA(Ab, lda, row_base, kk, lane);
#pragma unroll
    for (int j = 0; j < 4; j++) {
      v16h bf = load_fragB(Bb, ldb, col0 + j * 16, kk, lane);
      acc[j] = __builtin_amdgcn_wmma_f32_16x16x32_f16(false, a, false, bf, (short)0,
                                                      acc[j], false, false);
    }
  }
  int h = lane >> 4, c = lane & 15;
#pragma unroll
  for (int j = 0; j < 4; j++)
#pragma unroll
    for (int v = 0; v < 8; v++)
      Cb[(size_t)(row_base + v + 8 * h) * ldc + col0 + j * 16 + c] = (TOut)(acc[j][v] * scale);
}

// ---- Softmax over rows of 512 f32; writes f16 probs in place (row pitch stays 2048B) ----
__global__ void softmax_kernel(float* __restrict__ logits) {
  size_t row = blockIdx.x;
  float* src = logits + row * T_;
  _Float16* dst = (_Float16*)src;  // f16 probs occupy first half of each f32 row
  int tid = threadIdx.x;           // 128 threads * 4 = 512
  float v[4];
#pragma unroll
  for (int i = 0; i < 4; i++) v[i] = src[tid * 4 + i];
  float mx = fmaxf(fmaxf(v[0], v[1]), fmaxf(v[2], v[3]));
#pragma unroll
  for (int m = 1; m < 32; m <<= 1) mx = fmaxf(mx, __shfl_xor(mx, m, 32));
  __shared__ float red[4];
  if ((tid & 31) == 0) red[tid >> 5] = mx;
  __syncthreads();
  mx = fmaxf(fmaxf(red[0], red[1]), fmaxf(red[2], red[3]));
  float e[4], s = 0.0f;
#pragma unroll
  for (int i = 0; i < 4; i++) { e[i] = __expf(v[i] - mx); s += e[i]; }
#pragma unroll
  for (int m = 1; m < 32; m <<= 1) s += __shfl_xor(s, m, 32);
  __syncthreads();
  if ((tid & 31) == 0) red[tid >> 5] = s;
  __syncthreads();   // all reads of src complete before any f16 writes below
  float inv = 1.0f / (red[0] + red[1] + red[2] + red[3]);
#pragma unroll
  for (int i = 0; i < 4; i++) dst[tid * 4 + i] = (_Float16)(e[i] * inv);
}

// ---- Fused score GEMM + row argmax: idx[p] = argmax_q  q[p,:].pn[q,:] ----
// Two sequential half-passes of 8 column tiles each: 8 x v8f acc (64 VGPRs) + A frag
// + B frag + argmax state ~= 100 VGPRs -- fits without accumulator spills.
__global__ void __launch_bounds__(256, 1)
score_argmax_kernel(const _Float16* __restrict__ q,
                    const _Float16* __restrict__ pn,
                    int* __restrict__ idx_out) {
  int lane = threadIdx.x & 31, w = threadIdx.x >> 5;
  int b = blockIdx.y;
  int row_base = blockIdx.x * 16;
  const _Float16* qb = q + (size_t)b * P_ * D_;
  const _Float16* pb = pn + (size_t)b * P_ * D_;

  float bestv[8]; int besti[8];
#pragma unroll
  for (int v = 0; v < 8; v++) { bestv[v] = -3.4e38f; besti[v] = 0; }

  // wave w owns contiguous columns [w*256, (w+1)*256), processed in two halves of 128
#pragma unroll 1
  for (int half = 0; half < 2; half++) {
    int colw = w * 256 + half * 128;
    v8f acc[8];
#pragma unroll
    for (int j = 0; j < 8; j++) acc[j] = zero_v8f();
#pragma unroll 1
    for (int kk = 0; kk < D_; kk += 32) {
      v16h a = load_fragA(qb, D_, row_base, kk, lane);
#pragma unroll
      for (int j = 0; j < 8; j++) {
        v16h bf = load_fragB(pb, D_, colw + j * 16, kk, lane);
        acc[j] = __builtin_amdgcn_wmma_f32_16x16x32_f16(false, a, false, bf, (short)0,
                                                        acc[j], false, false);
      }
    }
    // fold into running argmax, ascending column order (first-index tie semantics)
#pragma unroll
    for (int j = 0; j < 8; j++) {
      int c = colw + j * 16 + (lane & 15);
#pragma unroll
      for (int v = 0; v < 8; v++) {
        float x = acc[j][v];
        if (x > bestv[v]) { bestv[v] = x; besti[v] = c; }
      }
    }
  }

  // reduce across 16 lanes of each half-wave (same output row per (v, lane>>4))
#pragma unroll
  for (int v = 0; v < 8; v++) {
#pragma unroll
    for (int m = 1; m < 16; m <<= 1) {
      float ov = __shfl_xor(bestv[v], m, 32);
      int   oi = __shfl_xor(besti[v], m, 32);
      if (ov > bestv[v] || (ov == bestv[v] && oi < besti[v])) { bestv[v] = ov; besti[v] = oi; }
    }
  }
  __shared__ float sv[8 * 16];
  __shared__ int   si[8 * 16];
  int h = lane >> 4;
  if ((lane & 15) == 0) {
#pragma unroll
    for (int v = 0; v < 8; v++) {
      int r = v + 8 * h;
      sv[w * 16 + r] = bestv[v];
      si[w * 16 + r] = besti[v];
    }
  }
  __syncthreads();
  if (threadIdx.x < 16) {
    int r = threadIdx.x;
    float bv = sv[r]; int bi = si[r];
    for (int ww = 1; ww < 8; ww++) {
      float ov = sv[ww * 16 + r]; int oi = si[ww * 16 + r];
      if (ov > bv || (ov == bv && oi < bi)) { bv = ov; bi = oi; }
    }
    idx_out[(size_t)b * P_ + row_base + r] = bi;
  }
}

// ---- Output assembly: tokens_out rows (cls copy | gathered patches | task copy) ----
__global__ void gather_tokens_kernel(const float* __restrict__ tokens,
                                     const int* __restrict__ idx,
                                     float* __restrict__ out) {
  int s = blockIdx.x, b = blockIdx.y, tid = threadIdx.x;
  size_t src_row;
  if (s == 0)            src_row = (size_t)b * S_;
  else if (s <= P_)      src_row = (size_t)b * S_ + 1 + idx[(size_t)b * P_ + (s - 1)];
  else                   src_row = (size_t)b * S_ + s;
  const float4* sp = (const float4*)(tokens + src_row * D_);
  float4* dp = (float4*)(out + ((size_t)b * S_ + s) * D_);
  dp[tid] = sp[tid];   // 256 threads * float4 = 1024 floats
}

__global__ void gather_meta_kernel(const int* __restrict__ pos,
                                   const int* __restrict__ mask,
                                   const int* __restrict__ idx,
                                   float* __restrict__ pos_out,
                                   float* __restrict__ mask_out) {
  int lin = blockIdx.x * blockDim.x + threadIdx.x;
  if (lin >= B_ * S_) return;
  int b = lin / S_, s = lin % S_;
  int src_col = (s == 0) ? 0 : (s <= P_ ? 1 + idx[b * P_ + (s - 1)] : s);
  pos_out[lin]  = (float)pos[(size_t)b * S_ + src_col];
  mask_out[lin] = (float)mask[(size_t)b * S_ + src_col];
}

extern "C" void kernel_launch(void* const* d_in, const int* in_sizes, int n_in,
                              void* d_out, int out_size, void* d_ws, size_t ws_size,
                              hipStream_t stream) {
  (void)in_sizes; (void)n_in; (void)out_size; (void)ws_size;
  const float* tokens = (const float*)d_in[0];
  const int*   pos    = (const int*)d_in[1];
  const int*   mask   = (const int*)d_in[2];
  float* out = (float*)d_out;

  char* ws = (char*)d_ws;
  size_t off = 0;
  _Float16* pn16   = (_Float16*)(ws + off); off += (size_t)B_ * P_ * D_ * 2;  // 33.5 MB
  _Float16* tn16   = (_Float16*)(ws + off); off += (size_t)B_ * T_ * D_ * 2;  //  8.4 MB
  _Float16* tnT16  = (_Float16*)(ws + off); off += (size_t)B_ * D_ * T_ * 2;  //  8.4 MB
  float*    logits = (float*)   (ws + off); off += (size_t)B_ * P_ * T_ * 4;  // 33.5 MB (probs f16 in place)
  _Float16* attn16 = (_Float16*)(ws + off); off += (size_t)B_ * P_ * D_ * 2;  // 33.5 MB
  _Float16* q16    = (_Float16*)(ws + off); off += (size_t)B_ * P_ * D_ * 2;  // 33.5 MB
  int*      idx    = (int*)     (ws + off); off += (size_t)B_ * P_ * 4;       // 64 KB

  const float scale = 0.03125f;  // 1/sqrt(1024)

  // 1) rms-norm patches and task tokens -> f16
  rmsnorm_kernel<float><<<dim3(P_, B_), 256, 0, stream>>>(tokens, pn16, S_, 1, P_);
  rmsnorm_kernel<float><<<dim3(T_, B_), 256, 0, stream>>>(tokens, tn16, S_, 1 + P_, T_);

  // 2) tnT = tn^T (makes probs@tn an NT GEMM)
  transpose_kernel<<<dim3((D_ * T_) / 256, B_), 256, 0, stream>>>(tn16, tnT16);

  // 3) logits = pn . tn^T * scale  (M=2048, N=512, K=1024)
  gemm_nt_kernel<float><<<dim3(P_ / 16, 1, B_), 256, 0, stream>>>(
      pn16, D_, tn16, D_, logits, T_, D_, scale,
      (size_t)P_ * D_, (size_t)T_ * D_, (size_t)P_ * T_);

  // 4) softmax rows of 512, f16 probs written in place (row pitch = 1024 halfs)
  softmax_kernel<<<dim3(B_ * P_), 128, 0, stream>>>(logits);

  // 5) attn = probs . tn  (NT with tnT: M=2048, N=1024, K=512), f16 output
  gemm_nt_kernel<_Float16><<<dim3(P_ / 16, 2, B_), 256, 0, stream>>>(
      (const _Float16*)logits, 1024, tnT16, T_, attn16, D_, T_, 1.0f,
      (size_t)P_ * 1024, (size_t)D_ * T_, (size_t)P_ * D_);

  // 6) q = rms-norm(attn)
  rmsnorm_kernel<_Float16><<<dim3(P_, B_), 256, 0, stream>>>(attn16, q16, P_, 0, P_);

  // 7) fused score GEMM (M=2048, N=2048, K=1024) + per-row argmax
  score_argmax_kernel<<<dim3(P_ / 16, B_), 256, 0, stream>>>(q16, pn16, idx);

  // 8) assemble outputs
  gather_tokens_kernel<<<dim3(S_, B_), 256, 0, stream>>>(tokens, idx, out);
  size_t tok_elems = (size_t)B_ * S_ * D_;
  gather_meta_kernel<<<dim3((B_ * S_ + 255) / 256), 256, 0, stream>>>(
      pos, mask, idx, out + tok_elems, out + tok_elems + (size_t)B_ * S_);
}
